// MultiHeadAttention_75479755260210
// MI455X (gfx1250) — compile-verified, matrix operand order checked by reading, not run on hardware
//
#include <hip/hip_runtime.h>
#include <hip/hip_bf16.h>

// ---------------------------------------------------------------------------
// Fused MHA block for MI455X (gfx1250, wave32).
// - All matmuls on v_wmma_f32_16x16x32_bf16 (bf16 operands, f32 accumulate)
// - Scores kernel tiles loaded via Tensor Data Mover (tensor_load_to_lds)
// - All other LDS fills vectorized (b128 copies / packed bf16 pair stores)
// ---------------------------------------------------------------------------

#define B_  8
#define S_  1024
#define D_  512
#define H_  8
#define DK_ 64
#define NEGV (-1e9f)

typedef __attribute__((ext_vector_type(16))) __bf16 v16bf;
typedef __attribute__((ext_vector_type(8)))  __bf16 v8bf;
typedef __attribute__((ext_vector_type(8)))  float  v8f;
typedef __attribute__((ext_vector_type(4)))  unsigned int u32x4;
typedef __attribute__((ext_vector_type(8)))  int i32x8;
typedef __attribute__((ext_vector_type(4)))  int i32x4;

union Frag { v16bf v; v8bf h[2]; };
union PkBf { unsigned int u; __bf16 b[2]; };

#if defined(__has_builtin)
#if __has_builtin(__builtin_amdgcn_tensor_load_to_lds) && __has_builtin(__builtin_amdgcn_s_wait_tensorcnt)
#define HAVE_TDM 1
#endif
#endif

#ifdef HAVE_TDM
// TDM load of a contiguous 64x64 bf16 tile (row stride 64 elems) into LDS.
// D# per CDNA5 ISA 08_async_tensor: group0={count,lds,global,type=2},
// group1={data_size=2B, dims/strides}, groups 2/3 zero (2-D tensor).
__device__ __forceinline__ void tdm_load_tile_64x64_bf16(const __bf16* gptr,
                                                         unsigned lds_off) {
    unsigned long long ga = (unsigned long long)(uintptr_t)gptr;
    u32x4 g0;
    g0[0] = 1u;                                            // count=1, user D#
    g0[1] = lds_off;                                       // LDS byte address
    g0[2] = (unsigned)(ga & 0xFFFFFFFFu);                  // global_addr lo
    g0[3] = (unsigned)((ga >> 32) & 0x01FFFFFFu) | (2u << 30); // addr hi|type=2
    i32x8 g1;
    g1[0] = (1 << 16);      // workgroup_mask=0, data_size=1 (2 bytes)
    g1[1] = (64 << 16);     // tensor_dim0[15:0]=64 (bits 63:48 of group)
    g1[2] = (64 << 16);     // tensor_dim0 hi=0 | tensor_dim1 lo=64
    g1[3] = (64 << 16);     // tensor_dim1 hi=0 | tile_dim0=64
    g1[4] = 64;             // tile_dim1=64, tile_dim2=0
    g1[5] = 64;             // tensor_dim0_stride lo = 64
    g1[6] = 0;              // stride0 hi | stride1 lo
    g1[7] = 0;              // stride1 hi
    i32x4 z4 = {0, 0, 0, 0};
#if __clang_major__ >= 23
    i32x8 z8 = {0, 0, 0, 0, 0, 0, 0, 0};
    __builtin_amdgcn_tensor_load_to_lds(g0, g1, z4, z4, z8, 0);
#else
    __builtin_amdgcn_tensor_load_to_lds(g0, g1, z4, z4, 0);
#endif
}
#endif

// ------------------------- weight transpose + cvt ---------------------------
// Wt[n][k] = (bf16) W[k][n]     (W is [K][N] row-major, K=N=512)
__global__ __launch_bounds__(256) void cvt_wt_kernel(const float* __restrict__ W,
                                                     __bf16* __restrict__ Wt) {
    int idx = blockIdx.x * 256 + threadIdx.x;          // 512*512 total
    int n = idx >> 9;
    int k = idx & 511;
    Wt[idx] = (__bf16)W[(size_t)k * 512 + n];
}

// ------------------------- QKV projection GEMM ------------------------------
// C[m][n] = A[m][:] . W[:][n] + bias[n]; A fp32 [8192][512], Wt bf16 [512][512]
// output bf16 in [B,H,S,DK] layout.
__global__ __launch_bounds__(128) void proj_gemm_kernel(
        const float* __restrict__ A, const __bf16* __restrict__ Wt,
        const float* __restrict__ bias, __bf16* __restrict__ Out) {
    __shared__ __align__(16) __bf16 As[64][32];
    __shared__ __align__(16) __bf16 Bs[64][32];
    const int tid  = threadIdx.x;
    const int wave = tid >> 5;
    const int lane = tid & 31;
    const int half = lane >> 4;
    const int lrow = lane & 15;
    const int m0 = blockIdx.x * 64;
    const int n0 = blockIdx.y * 64;

    v8f acc[4] = {};
    for (int kb = 0; kb < 512; kb += 32) {
        // A tile: fp32 -> packed bf16 pair per b32 store (1024 dwords)
        #pragma unroll
        for (int i = 0; i < 8; ++i) {
            int e = tid + i * 128;
            int r = e >> 4, c2 = e & 15;
            float2 f = *(const float2*)&A[(size_t)(m0 + r) * 512 + kb + c2 * 2];
            PkBf pk; pk.b[0] = (__bf16)f.x; pk.b[1] = (__bf16)f.y;
            *(unsigned int*)&As[r][c2 * 2] = pk.u;
        }
        // B tile: bf16 source, 16-byte chunk copies (256 chunks)
        #pragma unroll
        for (int i = 0; i < 2; ++i) {
            int e = tid + i * 128;
            int r = e >> 2, c = (e & 3) * 8;
            *(uint4*)&Bs[r][c] = *(const uint4*)&Wt[(size_t)(n0 + r) * 512 + kb + c];
        }
        __syncthreads();
        Frag a;
        const int row = wave * 16 + lrow;
        a.h[0] = *(const v8bf*)&As[row][half * 8];
        a.h[1] = *(const v8bf*)&As[row][16 + half * 8];
        #pragma unroll
        for (int j = 0; j < 4; ++j) {
            Frag b;
            const int col = j * 16 + lrow;
            b.h[0] = *(const v8bf*)&Bs[col][half * 8];
            b.h[1] = *(const v8bf*)&Bs[col][16 + half * 8];
            acc[j] = __builtin_amdgcn_wmma_f32_16x16x32_bf16(
                false, a.v, false, b.v, (short)0, acc[j], false, false);
        }
        __syncthreads();
    }
    #pragma unroll
    for (int j = 0; j < 4; ++j) {
        #pragma unroll
        for (int r = 0; r < 8; ++r) {
            int m = m0 + wave * 16 + r + 8 * half;
            int n = n0 + j * 16 + lrow;
            float val = acc[j][r] + bias[n];
            int b = m >> 10, s = m & 1023;
            int h = n >> 6,  d = n & 63;
            Out[((((size_t)b * H_ + h) << 10) + s) * DK_ + d] = (__bf16)val;
        }
    }
}

// ------------------------- scores = q k^T * scale, mask, +adjoin ------------
// per (b,h): q,k bf16 [S][64]; q/k 64x64 tiles pulled into LDS by the TDM.
__global__ __launch_bounds__(128) void scores_kernel(
        const __bf16* __restrict__ qb, const __bf16* __restrict__ kb,
        const int* __restrict__ mask, const float* __restrict__ adjoin,
        float* __restrict__ attn) {
    __shared__ __align__(16) __bf16 As[64][64];
    __shared__ __align__(16) __bf16 Bs[64][64];
    const int tid  = threadIdx.x;
    const int wave = tid >> 5;
    const int lane = tid & 31;
    const int half = lane >> 4;
    const int lrow = lane & 15;
    const int m0 = blockIdx.x * 64;
    const int n0 = blockIdx.y * 64;
    const int bh = blockIdx.z;
    const int bb = bh >> 3;
    const size_t base = (size_t)bh * S_ * DK_;
    const __bf16* qtile = qb + base + (size_t)m0 * DK_;   // contiguous 8 KB
    const __bf16* ktile = kb + base + (size_t)n0 * DK_;   // contiguous 8 KB

#ifdef HAVE_TDM
    if (tid < 32) {   // one wave issues both DMA tile loads
        tdm_load_tile_64x64_bf16(qtile, (unsigned)(uintptr_t)&As[0][0]);
        tdm_load_tile_64x64_bf16(ktile, (unsigned)(uintptr_t)&Bs[0][0]);
        __builtin_amdgcn_s_wait_tensorcnt(0);
    }
#else
    #pragma unroll
    for (int i = 0; i < 4; ++i) {       // 512 16-byte chunks per tile
        int e = tid + i * 128;
        int r = e >> 3, c = (e & 7) * 8;
        *(uint4*)&As[r][c] = *(const uint4*)&qtile[(size_t)r * DK_ + c];
        *(uint4*)&Bs[r][c] = *(const uint4*)&ktile[(size_t)r * DK_ + c];
    }
#endif
    __syncthreads();

    v8f acc[4] = {};
    const int row = wave * 16 + lrow;
    #pragma unroll
    for (int kv = 0; kv < DK_; kv += 32) {
        Frag a;
        a.h[0] = *(const v8bf*)&As[row][kv + half * 8];
        a.h[1] = *(const v8bf*)&As[row][kv + 16 + half * 8];
        #pragma unroll
        for (int j = 0; j < 4; ++j) {
            Frag b;
            const int col = j * 16 + lrow;
            b.h[0] = *(const v8bf*)&Bs[col][kv + half * 8];
            b.h[1] = *(const v8bf*)&Bs[col][kv + 16 + half * 8];
            acc[j] = __builtin_amdgcn_wmma_f32_16x16x32_bf16(
                false, a.v, false, b.v, (short)0, acc[j], false, false);
        }
    }
    const float scale = 0.125f;   // 1/sqrt(64)
    #pragma unroll
    for (int j = 0; j < 4; ++j) {
        #pragma unroll
        for (int r = 0; r < 8; ++r) {
            int m = m0 + wave * 16 + r + 8 * half;
            int n = n0 + j * 16 + lrow;
            float val = acc[j][r] * scale;
            size_t mi = ((size_t)bb * S_ + m) * S_ + n;   // [B,S,S]
            if (mask[mi] != 0) val = NEGV;                // masked_fill first
            val += adjoin[mi];                            // then adjoin bias
            attn[(size_t)bh * S_ * S_ + (size_t)m * S_ + n] = val;
        }
    }
}

// ------------------------- softmax rows (in place) --------------------------
__global__ __launch_bounds__(256) void softmax_kernel(float* __restrict__ attn) {
    __shared__ float red[256];
    const int tid = threadIdx.x;
    float* p = attn + (size_t)blockIdx.x * S_;
    float4 x = ((float4*)p)[tid];
    float mx = fmaxf(fmaxf(x.x, x.y), fmaxf(x.z, x.w));
    red[tid] = mx; __syncthreads();
    #pragma unroll
    for (int s = 128; s > 0; s >>= 1) {
        if (tid < s) red[tid] = fmaxf(red[tid], red[tid + s]);
        __syncthreads();
    }
    mx = red[0]; __syncthreads();
    x.x = __expf(x.x - mx); x.y = __expf(x.y - mx);
    x.z = __expf(x.z - mx); x.w = __expf(x.w - mx);
    red[tid] = x.x + x.y + x.z + x.w; __syncthreads();
    #pragma unroll
    for (int s = 128; s > 0; s >>= 1) {
        if (tid < s) red[tid] += red[tid + s];
        __syncthreads();
    }
    float inv = 1.0f / red[0];
    x.x *= inv; x.y *= inv; x.z *= inv; x.w *= inv;
    ((float4*)p)[tid] = x;
}

// ------------------------- ctx = attn @ v -----------------------------------
// per (b,h): attn fp32 [S][S] (packed to bf16 on LDS fill), v bf16 [S][64].
// output bf16 ctx in [B,S,H*DV] = [8192][512] layout.
__global__ __launch_bounds__(128) void ctx_gemm_kernel(
        const float* __restrict__ attn, const __bf16* __restrict__ vb,
        __bf16* __restrict__ ctx) {
    __shared__ __align__(16) __bf16 As[64][32];
    __shared__ __align__(16) __bf16 Bs[64][32];
    const int tid  = threadIdx.x;
    const int wave = tid >> 5;
    const int lane = tid & 31;
    const int half = lane >> 4;
    const int lrow = lane & 15;
    const int m0 = blockIdx.x * 64;
    const int bh = blockIdx.y;
    const size_t abase = (size_t)bh * S_ * S_;
    const size_t vbase = (size_t)bh * S_ * DK_;

    v8f acc[4] = {};
    for (int kv = 0; kv < S_; kv += 32) {
        // attn tile: fp32 -> packed bf16 pairs
        #pragma unroll
        for (int i = 0; i < 8; ++i) {
            int e = tid + i * 128;
            int r = e >> 4, c2 = e & 15;
            float2 f = *(const float2*)&attn[abase + (size_t)(m0 + r) * S_ + kv + c2 * 2];
            PkBf pk; pk.b[0] = (__bf16)f.x; pk.b[1] = (__bf16)f.y;
            *(unsigned int*)&As[r][c2 * 2] = pk.u;
        }
        // v tile transposed (Bs[n][k] = v[k][n]), packed pair per b32 store
        #pragma unroll
        for (int i = 0; i < 8; ++i) {
            int e = tid + i * 128;
            int r = e >> 4, c2 = e & 15;           // r = n, c2 = k pair
            PkBf pk;
            pk.b[0] = vb[vbase + (size_t)(kv + c2 * 2)     * DK_ + r];
            pk.b[1] = vb[vbase + (size_t)(kv + c2 * 2 + 1) * DK_ + r];
            *(unsigned int*)&Bs[r][c2 * 2] = pk.u;
        }
        __syncthreads();
        Frag a;
        const int row = wave * 16 + lrow;
        a.h[0] = *(const v8bf*)&As[row][half * 8];
        a.h[1] = *(const v8bf*)&As[row][16 + half * 8];
        #pragma unroll
        for (int j = 0; j < 4; ++j) {
            Frag b;
            const int col = j * 16 + lrow;
            b.h[0] = *(const v8bf*)&Bs[col][half * 8];
            b.h[1] = *(const v8bf*)&Bs[col][16 + half * 8];
            acc[j] = __builtin_amdgcn_wmma_f32_16x16x32_bf16(
                false, a.v, false, b.v, (short)0, acc[j], false, false);
        }
        __syncthreads();
    }
    const int bb = bh >> 3, hh = bh & 7;
    #pragma unroll
    for (int j = 0; j < 4; ++j) {
        #pragma unroll
        for (int r = 0; r < 8; ++r) {
            int m = m0 + wave * 16 + r + 8 * half;   // s index
            int n = j * 16 + lrow;                   // d index (0..63)
            ctx[((size_t)bb * S_ + m) * D_ + hh * DK_ + n] = (__bf16)acc[j][r];
        }
    }
}

// ------------------------- out = ctx @ Wo + bo + Q --------------------------
__global__ __launch_bounds__(128) void out_gemm_kernel(
        const __bf16* __restrict__ ctx, const __bf16* __restrict__ WoT,
        const float* __restrict__ bo, const float* __restrict__ Q,
        float* __restrict__ pre) {
    __shared__ __align__(16) __bf16 As[64][32];
    __shared__ __align__(16) __bf16 Bs[64][32];
    const int tid  = threadIdx.x;
    const int wave = tid >> 5;
    const int lane = tid & 31;
    const int half = lane >> 4;
    const int lrow = lane & 15;
    const int m0 = blockIdx.x * 64;
    const int n0 = blockIdx.y * 64;

    v8f acc[4] = {};
    for (int kb = 0; kb < 512; kb += 32) {
        #pragma unroll
        for (int i = 0; i < 2; ++i) {
            int e = tid + i * 128;
            int r = e >> 2, c = (e & 3) * 8;
            *(uint4*)&As[r][c] = *(const uint4*)&ctx[(size_t)(m0 + r) * 512 + kb + c];
            *(uint4*)&Bs[r][c] = *(const uint4*)&WoT[(size_t)(n0 + r) * 512 + kb + c];
        }
        __syncthreads();
        Frag a;
        const int row = wave * 16 + lrow;
        a.h[0] = *(const v8bf*)&As[row][half * 8];
        a.h[1] = *(const v8bf*)&As[row][16 + half * 8];
        #pragma unroll
        for (int j = 0; j < 4; ++j) {
            Frag b;
            const int col = j * 16 + lrow;
            b.h[0] = *(const v8bf*)&Bs[col][half * 8];
            b.h[1] = *(const v8bf*)&Bs[col][16 + half * 8];
            acc[j] = __builtin_amdgcn_wmma_f32_16x16x32_bf16(
                false, a.v, false, b.v, (short)0, acc[j], false, false);
        }
        __syncthreads();
    }
    #pragma unroll
    for (int j = 0; j < 4; ++j) {
        #pragma unroll
        for (int r = 0; r < 8; ++r) {
            int m = m0 + wave * 16 + r + 8 * half;
            int n = n0 + j * 16 + lrow;
            size_t oi = (size_t)m * 512 + n;
            pre[oi] = acc[j][r] + bo[n] + Q[oi];      // residual = original Q
        }
    }
}

// ------------------------- LayerNorm ----------------------------------------
__global__ __launch_bounds__(256) void layernorm_kernel(
        const float* __restrict__ pre, const float* __restrict__ g,
        const float* __restrict__ bta, float* __restrict__ out) {
    __shared__ float red[256];
    const int tid = threadIdx.x;
    const float* p = pre + (size_t)blockIdx.x * D_;
    float2 x = ((const float2*)p)[tid];
    red[tid] = x.x + x.y; __syncthreads();
    #pragma unroll
    for (int s = 128; s > 0; s >>= 1) {
        if (tid < s) red[tid] += red[tid + s];
        __syncthreads();
    }
    float mu = red[0] * (1.0f / D_); __syncthreads();
    float dx = x.x - mu, dy = x.y - mu;
    red[tid] = dx * dx + dy * dy; __syncthreads();
    #pragma unroll
    for (int s = 128; s > 0; s >>= 1) {
        if (tid < s) red[tid] += red[tid + s];
        __syncthreads();
    }
    float inv = rsqrtf(red[0] * (1.0f / D_) + 1e-6f);
    float2 gg = ((const float2*)g)[tid];
    float2 bb = ((const float2*)bta)[tid];
    float2 o;
    o.x = dx * inv * gg.x + bb.x;
    o.y = dy * inv * gg.y + bb.y;
    ((float2*)(out + (size_t)blockIdx.x * D_))[tid] = o;
}

// ---------------------------------------------------------------------------
extern "C" void kernel_launch(void* const* d_in, const int* in_sizes, int n_in,
                              void* d_out, int out_size, void* d_ws, size_t ws_size,
                              hipStream_t stream) {
    const float* Q      = (const float*)d_in[0];
    const float* K      = (const float*)d_in[1];
    const float* V      = (const float*)d_in[2];
    const int*   mask   = (const int*)  d_in[3];
    const float* adjoin = (const float*)d_in[4];
    const float* Wq     = (const float*)d_in[5];
    const float* bq     = (const float*)d_in[6];
    const float* Wk     = (const float*)d_in[7];
    const float* bk     = (const float*)d_in[8];
    const float* Wv     = (const float*)d_in[9];
    const float* bv     = (const float*)d_in[10];
    const float* Wo     = (const float*)d_in[11];
    const float* bo     = (const float*)d_in[12];
    const float* ln_g   = (const float*)d_in[13];
    const float* ln_b   = (const float*)d_in[14];

    float* out  = (float*)d_out;                       // [B,S,D]
    float* attn = out + (size_t)B_ * S_ * D_;          // [B,H,S,S]

    // workspace layout
    char* ws = (char*)d_ws;
    const size_t SZ_HEAD = (size_t)B_ * H_ * S_ * DK_ * sizeof(__bf16); // 8 MB
    const size_t SZ_W    = (size_t)512 * 512 * sizeof(__bf16);          // 0.5 MB
    __bf16* qb   = (__bf16*)(ws);
    __bf16* kb   = (__bf16*)(ws + SZ_HEAD);
    __bf16* vb   = (__bf16*)(ws + 2 * SZ_HEAD);
    __bf16* ctx  = (__bf16*)(ws + 3 * SZ_HEAD);                         // 8 MB
    __bf16* WqT  = (__bf16*)(ws + 4 * SZ_HEAD);
    __bf16* WkT  = (__bf16*)(ws + 4 * SZ_HEAD + SZ_W);
    __bf16* WvT  = (__bf16*)(ws + 4 * SZ_HEAD + 2 * SZ_W);
    __bf16* WoT  = (__bf16*)(ws + 4 * SZ_HEAD + 3 * SZ_W);
    float*  pre  = (float*) (ws + 4 * SZ_HEAD + 4 * SZ_W);              // 16 MB

    // 1. weight transpose + bf16 convert
    cvt_wt_kernel<<<1024, 256, 0, stream>>>(Wq, WqT);
    cvt_wt_kernel<<<1024, 256, 0, stream>>>(Wk, WkT);
    cvt_wt_kernel<<<1024, 256, 0, stream>>>(Wv, WvT);
    cvt_wt_kernel<<<1024, 256, 0, stream>>>(Wo, WoT);

    // 2. QKV projections -> [B,H,S,64] bf16
    dim3 gproj(128, 8);
    proj_gemm_kernel<<<gproj, 128, 0, stream>>>(Q, WqT, bq, qb);
    proj_gemm_kernel<<<gproj, 128, 0, stream>>>(K, WkT, bk, kb);
    proj_gemm_kernel<<<gproj, 128, 0, stream>>>(V, WvT, bv, vb);

    // 3. scores (+ scale, mask, adjoin) -> attn region of d_out (TDM tiles)
    dim3 gsc(16, 16, 64);
    scores_kernel<<<gsc, 128, 0, stream>>>(qb, kb, mask, adjoin, attn);

    // 4. softmax in place
    softmax_kernel<<<B_ * H_ * S_, 256, 0, stream>>>(attn);

    // 5. ctx = attn @ v -> [8192][512] bf16
    dim3 gctx(16, 64);
    ctx_gemm_kernel<<<gctx, 128, 0, stream>>>(attn, vb, ctx);

    // 6. out projection + bias + residual Q -> pre-LN fp32
    out_gemm_kernel<<<gproj, 128, 0, stream>>>(ctx, WoT, bo, Q, pre);

    // 7. LayerNorm -> d_out
    layernorm_kernel<<<B_ * S_, 256, 0, stream>>>(pre, ln_g, ln_b, out);
}
